// MultiHeadAttention_83863531421946
// MI455X (gfx1250) — matrix operand order repaired and checked by reading, not compiled, not run on hardware
//
#include <hip/hip_runtime.h>
#include <hip/hip_bf16.h>

// ---------------------------------------------------------------------------
// MHA forward for MI455X (gfx1250, wave32, WMMA + Tensor Data Mover).
// fp32 -> bf16 once, all contractions via v_wmma_f32_16x16x32_bf16 (fp32 acc),
// softmax in fp32 VALU. Flash-attention K/V tiles double-buffered in LDS and
// prefetched by the TDM (tensor_load_to_lds, TENSORcnt pipelined: issue next
// tile, wait <=2 so the older pair has landed).
// ---------------------------------------------------------------------------

#define DMODEL 1024
#define HEADS  16
#define DK     64
#define BATCH  4
#define SLEN   2048
#define MTOT   (BATCH * SLEN)   // 8192

typedef __bf16 v16bf __attribute__((ext_vector_type(16)));
typedef float  v8f   __attribute__((ext_vector_type(8)));

typedef unsigned int tdm_u4 __attribute__((ext_vector_type(4)));
typedef int          tdm_i8 __attribute__((ext_vector_type(8)));
typedef int          tdm_i4 __attribute__((ext_vector_type(4)));

union Frag16 { v16bf v; uint4 u[2]; };

__device__ __forceinline__ __bf16 f2bf(float f) {
    unsigned u = __builtin_bit_cast(unsigned, f);
    unsigned r = (u + 0x7FFFu + ((u >> 16) & 1u)) >> 16;   // RNE
    unsigned short h = (unsigned short)r;
    return __builtin_bit_cast(__bf16, h);
}

// ---------------------------------------------------------------------------
// TDM: DMA a 2D bf16 tile [rows x 64] (row stride `stride_elems`) from global
// into LDS via Tensor DMA Descriptor (D#) per CDNA5 ISA.
//   group0: count=1 | lds_addr | global_addr(57b) | type=2
//   group1: data_size=1(2B), tensor_dim0=64, tensor_dim1=rows,
//           tile_dim0=64, tile_dim1=rows, tensor_dim0_stride=stride_elems
// Wave-level op; tracked by TENSORcnt (in-order per wave).
// ---------------------------------------------------------------------------
__device__ __forceinline__ void tdm_load_2d(const __bf16* gptr, void* lds_ptr,
                                            int rows, int stride_elems) {
    unsigned long long ga = (unsigned long long)(uintptr_t)gptr;
    unsigned lds = (unsigned)(uintptr_t)lds_ptr;   // low 32b = LDS byte offset

    tdm_u4 g0;
    g0.x = 0x1u;                                        // count=1, no gather
    g0.y = lds;                                         // lds_addr
    g0.z = (unsigned)(ga & 0xFFFFFFFFu);                // global_addr[31:0]
    g0.w = (unsigned)((ga >> 32) & 0x01FFFFFFu)         // global_addr[56:32]
         | (2u << 30);                                  // type=2 ("image")

    tdm_i8 g1;
    g1[0] = 0x00010000;                                 // data_size=1 (2 bytes)
    g1[1] = (int)(64u << 16);                           // tensor_dim0 = 64
    g1[2] = (int)(((unsigned)rows) << 16);              // tensor_dim1 = rows
    g1[3] = (int)(64u << 16);                           // tile_dim0 = 64
    g1[4] = rows;                                       // tile_dim1 = rows
    g1[5] = stride_elems;                               // tensor_dim0_stride
    g1[6] = 0;
    g1[7] = 0;

    tdm_i4 z4 = {0, 0, 0, 0};
#if defined(__clang_major__) && (__clang_major__ >= 23)
    tdm_i8 z8 = {0, 0, 0, 0, 0, 0, 0, 0};
    __builtin_amdgcn_tensor_load_to_lds(g0, g1, z4, z4, z8, 0);
#else
    __builtin_amdgcn_tensor_load_to_lds(g0, g1, z4, z4, 0);
#endif
}

// ---------------------------------------------------------------------------
// fp32 -> bf16 elementwise convert
// ---------------------------------------------------------------------------
__global__ void k_cvt(const float* __restrict__ src, __bf16* __restrict__ dst, int n) {
    int i = blockIdx.x * blockDim.x + threadIdx.x;
    int stride = gridDim.x * blockDim.x;
    for (; i < n; i += stride) dst[i] = f2bf(src[i]);
}

// ---------------------------------------------------------------------------
// C = X (MTOT x 1024, bf16) * W^T (W is [1024 out, 1024 in], bf16)
// MODE 0: write bf16 head-split [B,H,S,DK];  MODE 1: write fp32 [MTOT,1024]
// 256 thr = 8 waves (4M x 2N); block tile 64 x 128; wave tile 16 x 64.
// ---------------------------------------------------------------------------
template <int MODE>
__global__ __launch_bounds__(256) void k_gemm(const __bf16* __restrict__ X,
                                              const __bf16* __restrict__ W,
                                              void* __restrict__ dstv) {
    const int tid   = threadIdx.x;
    const int wave  = tid >> 5;
    const int lane  = tid & 31;
    const int lhalf = lane >> 4;
    const int l16   = lane & 15;
    const int wm    = wave & 3;
    const int wn    = wave >> 2;
    const int m0    = blockIdx.x * 64  + wm * 16;
    const int n0    = blockIdx.y * 128 + wn * 64;

    const __bf16* xr = X + (size_t)(m0 + l16) * DMODEL;

    v8f acc[4] = {};
    for (int kk = 0; kk < DMODEL; kk += 32) {
        Frag16 a;
        const int ka = kk + lhalf * 16;
        a.u[0] = *reinterpret_cast<const uint4*>(xr + ka);
        a.u[1] = *reinterpret_cast<const uint4*>(xr + ka + 8);
#pragma unroll
        for (int t = 0; t < 4; ++t) {
            Frag16 bf;
            const __bf16* wr = W + (size_t)(n0 + t * 16 + l16) * DMODEL + kk + lhalf * 16;
            bf.u[0] = *reinterpret_cast<const uint4*>(wr);
            bf.u[1] = *reinterpret_cast<const uint4*>(wr + 8);
            acc[t] = __builtin_amdgcn_wmma_f32_16x16x32_bf16(
                false, a.v, false, bf.v, (short)0, acc[t], false, false);
        }
    }

    if (MODE == 0) {
        __bf16* dst = (__bf16*)dstv;
#pragma unroll
        for (int t = 0; t < 4; ++t)
#pragma unroll
            for (int r = 0; r < 8; ++r) {
                const int m = m0 + r + 8 * lhalf;
                const int n = n0 + t * 16 + l16;
                const int bb = m >> 11, ss = m & (SLEN - 1);
                const int hh = n >> 6,  dd = n & 63;
                dst[((size_t)(bb * HEADS + hh) * SLEN + ss) * DK + dd] = f2bf(acc[t][r]);
            }
    } else {
        float* dst = (float*)dstv;
#pragma unroll
        for (int t = 0; t < 4; ++t)
#pragma unroll
            for (int r = 0; r < 8; ++r) {
                const int m = m0 + r + 8 * lhalf;
                const int n = n0 + t * 16 + l16;
                dst[(size_t)m * DMODEL + n] = acc[t][r];
            }
    }
}

// ---------------------------------------------------------------------------
// Fused flash attention per (b,h). Block = 128 threads = 4 waves.
// Double-buffered K/V tiles prefetched by the TDM: wave 0 issues next tile's
// tensor_load_to_lds pair, then s_wait_tensorcnt(2) so the current tile's
// (older, in-order) pair has landed. B-fragments preloaded in bulk so the
// ds_load_b128s clause and WMMAs issue back-to-back.
// ---------------------------------------------------------------------------
__global__ __launch_bounds__(128) void k_flash(const __bf16* __restrict__ qh,
                                               const __bf16* __restrict__ kh,
                                               const __bf16* __restrict__ vh,
                                               __bf16* __restrict__ ctx) {
    const int bh    = blockIdx.y;
    const int b     = bh / HEADS;
    const int h     = bh % HEADS;
    const int qtile = blockIdx.x * 64;
    const int tid   = threadIdx.x;
    const int wave  = tid >> 5;
    const int lane  = tid & 31;
    const int lhalf = lane >> 4;
    const int l16   = lane & 15;

    __shared__ __align__(16) __bf16 kT[2][64][64];   // K tiles [j][d] (TDM dest)
    __shared__ __align__(16) __bf16 vR[2][64][64];   // V tiles [j][d] (TDM dest)
    __shared__ __align__(16) __bf16 vT[DK][72];      // V tile transposed [d][j]
    __shared__ __align__(16) __bf16 Pl[4][16][72];   // per-wave probs [row][j]

    const __bf16* qbase = qh + (size_t)bh * SLEN * DK;
    const __bf16* kbase = kh + (size_t)bh * SLEN * DK;
    const __bf16* vbase = vh + (size_t)bh * SLEN * DK;

    // Q A-fragments live in registers for the whole K loop (d_k=64 -> 2 frags)
    Frag16 aq[2];
    {
        const int row = qtile + wave * 16 + l16;
        const __bf16* qr = qbase + (size_t)row * DK;
#pragma unroll
        for (int kc = 0; kc < 2; ++kc) {
            const int k0 = kc * 32 + lhalf * 16;
            aq[kc].u[0] = *reinterpret_cast<const uint4*>(qr + k0);
            aq[kc].u[1] = *reinterpret_cast<const uint4*>(qr + k0 + 8);
        }
    }

    v8f o[4] = {};
    float mrow[8], lrow[8];
#pragma unroll
    for (int r = 0; r < 8; ++r) { mrow[r] = -3.0e38f; lrow[r] = 0.0f; }

    // Prologue: TDM prefetch of the first K/V tiles into buffer 0
    if (wave == 0) {
        tdm_load_2d(kbase, &kT[0][0][0], 64, DK);
        tdm_load_2d(vbase, &vR[0][0][0], 64, DK);
    }

    int cur = 0;
    for (int j0 = 0; j0 < SLEN; j0 += 64) {
        // --- pipeline: issue next tile's DMA, then drain current tile ---
        if (wave == 0) {
            if (j0 + 64 < SLEN) {
                tdm_load_2d(kbase + (size_t)(j0 + 64) * DK, &kT[cur ^ 1][0][0], 64, DK);
                tdm_load_2d(vbase + (size_t)(j0 + 64) * DK, &vR[cur ^ 1][0][0], 64, DK);
                __builtin_amdgcn_s_wait_tensorcnt(2);  // older pair (current tile) done
            } else {
                __builtin_amdgcn_s_wait_tensorcnt(0);
            }
        }
        __syncthreads();   // current tile visible; prev iteration's readers done

        // --- transpose V inside LDS: vR[cur][j][d] -> vT[d][j] ---
#pragma unroll
        for (int c = 0; c < 4; ++c) {
            const int chunk = tid * 4 + c;        // 512 chunks of 8 bf16
            const int j  = chunk >> 3;
            const int d8 = (chunk & 7) * 8;
            uint4 raw = *reinterpret_cast<const uint4*>(&vR[cur][j][d8]);
            const __bf16* e = reinterpret_cast<const __bf16*>(&raw);
#pragma unroll
            for (int t = 0; t < 8; ++t) vT[d8 + t][j] = e[t];
        }
        __syncthreads();

        // --- scores: preload all 4 B-fragments, then WMMA burst ---
        Frag16 bk[4][2];
#pragma unroll
        for (int jt = 0; jt < 4; ++jt) {
            const __bf16* kr = &kT[cur][jt * 16 + l16][0];
#pragma unroll
            for (int kc = 0; kc < 2; ++kc) {
                const int d0 = kc * 32 + lhalf * 16;
                bk[jt][kc].u[0] = *reinterpret_cast<const uint4*>(kr + d0);
                bk[jt][kc].u[1] = *reinterpret_cast<const uint4*>(kr + d0 + 8);
            }
        }
        v8f s[4];
#pragma unroll
        for (int jt = 0; jt < 4; ++jt) {
            v8f accs = {};
            accs = __builtin_amdgcn_wmma_f32_16x16x32_bf16(
                false, aq[0].v, false, bk[jt][0].v, (short)0, accs, false, false);
            accs = __builtin_amdgcn_wmma_f32_16x16x32_bf16(
                false, aq[1].v, false, bk[jt][1].v, (short)0, accs, false, false);
            s[jt] = accs;
        }

        // --- online softmax (rows spread over the 16-lane halves) ---
        float alpha[8];
#pragma unroll
        for (int r = 0; r < 8; ++r) {
            float x0 = s[0][r] * 0.125f, x1 = s[1][r] * 0.125f;
            float x2 = s[2][r] * 0.125f, x3 = s[3][r] * 0.125f;
            float m = fmaxf(fmaxf(x0, x1), fmaxf(x2, x3));
            for (int msk = 8; msk >= 1; msk >>= 1) m = fmaxf(m, __shfl_xor(m, msk, 32));
            const float mnew = fmaxf(mrow[r], m);
            alpha[r] = __expf(mrow[r] - mnew);
            mrow[r] = mnew;
            float p0 = __expf(x0 - mnew), p1 = __expf(x1 - mnew);
            float p2 = __expf(x2 - mnew), p3 = __expf(x3 - mnew);
            s[0][r] = p0; s[1][r] = p1; s[2][r] = p2; s[3][r] = p3;
            float ls = p0 + p1 + p2 + p3;
            for (int msk = 8; msk >= 1; msk >>= 1) ls += __shfl_xor(ls, msk, 32);
            lrow[r] = lrow[r] * alpha[r] + ls;
#pragma unroll
            for (int dt = 0; dt < 4; ++dt) o[dt][r] *= alpha[r];
        }

        // --- P (C/D layout) -> LDS -> A-fragment layout ---
#pragma unroll
        for (int jt = 0; jt < 4; ++jt)
#pragma unroll
            for (int r = 0; r < 8; ++r)
                Pl[wave][r + 8 * lhalf][jt * 16 + l16] = f2bf(s[jt][r]);
        __syncthreads();   // also: all kT[cur] readers done before next TDM issue

        Frag16 pa[2];
        {
            const __bf16* pr = &Pl[wave][l16][0];
#pragma unroll
            for (int kc = 0; kc < 2; ++kc) {
                const int k0 = kc * 32 + lhalf * 16;
                pa[kc].u[0] = *reinterpret_cast<const uint4*>(pr + k0);
                pa[kc].u[1] = *reinterpret_cast<const uint4*>(pr + k0 + 8);
            }
        }

        // --- O += P * V: preload all 4 B-fragments, then WMMA burst ---
        Frag16 bv[4][2];
#pragma unroll
        for (int dt = 0; dt < 4; ++dt) {
            const __bf16* vr = &vT[dt * 16 + l16][0];
#pragma unroll
            for (int kc = 0; kc < 2; ++kc) {
                const int jl = kc * 32 + lhalf * 16;
                bv[dt][kc].u[0] = *reinterpret_cast<const uint4*>(vr + jl);
                bv[dt][kc].u[1] = *reinterpret_cast<const uint4*>(vr + jl + 8);
            }
        }
#pragma unroll
        for (int dt = 0; dt < 4; ++dt) {
            o[dt] = __builtin_amdgcn_wmma_f32_16x16x32_bf16(
                false, pa[0].v, false, bv[dt][0].v, (short)0, o[dt], false, false);
            o[dt] = __builtin_amdgcn_wmma_f32_16x16x32_bf16(
                false, pa[1].v, false, bv[dt][1].v, (short)0, o[dt], false, false);
        }

        cur ^= 1;
    }

    // --- normalize, merge heads: ctx[b, s, h*64 + d] (bf16) ---
#pragma unroll
    for (int r = 0; r < 8; ++r) {
        const float inv = 1.0f / lrow[r];
        const int srow = qtile + wave * 16 + r + 8 * lhalf;
        __bf16* dst = ctx + ((size_t)(b * SLEN + srow) * DMODEL) + h * DK;
#pragma unroll
        for (int dt = 0; dt < 4; ++dt)
            dst[dt * 16 + l16] = f2bf(o[dt][r] * inv);
    }
}

// ---------------------------------------------------------------------------
// Host-side launcher
// ---------------------------------------------------------------------------
extern "C" void kernel_launch(void* const* d_in, const int* in_sizes, int n_in,
                              void* d_out, int out_size, void* d_ws, size_t ws_size,
                              hipStream_t stream) {
    (void)in_sizes; (void)n_in; (void)out_size; (void)ws_size;

    const float* Q  = (const float*)d_in[0];
    const float* K  = (const float*)d_in[1];
    const float* V  = (const float*)d_in[2];
    const float* Wq = (const float*)d_in[3];
    const float* Wk = (const float*)d_in[4];
    const float* Wv = (const float*)d_in[5];
    const float* Wo = (const float*)d_in[6];

    const int NX = MTOT * DMODEL;      // 8,388,608
    const int NW = DMODEL * DMODEL;    // 1,048,576

    char* ws = (char*)d_ws;
    size_t off = 0;
    auto take = [&](size_t bytes) -> char* {
        char* p = ws + off;
        off = (off + bytes + 255) & ~(size_t)255;
        return p;
    };

    __bf16* Qb  = (__bf16*)take((size_t)NX * 2);
    __bf16* Kb  = (__bf16*)take((size_t)NX * 2);
    __bf16* Vb  = (__bf16*)take((size_t)NX * 2);
    __bf16* Wqb = (__bf16*)take((size_t)NW * 2);
    __bf16* Wkb = (__bf16*)take((size_t)NW * 2);
    __bf16* Wvb = (__bf16*)take((size_t)NW * 2);
    __bf16* Wob = (__bf16*)take((size_t)NW * 2);
    __bf16* qhb = (__bf16*)take((size_t)NX * 2);   // [B,H,S,DK]
    __bf16* khb = (__bf16*)take((size_t)NX * 2);
    __bf16* vhb = (__bf16*)take((size_t)NX * 2);
    __bf16* ctx = (__bf16*)take((size_t)NX * 2);   // [B,S,DMODEL]

    // 1) fp32 -> bf16
    k_cvt<<<2048, 256, 0, stream>>>(Q,  Qb,  NX);
    k_cvt<<<2048, 256, 0, stream>>>(K,  Kb,  NX);
    k_cvt<<<2048, 256, 0, stream>>>(V,  Vb,  NX);
    k_cvt<<<1024, 256, 0, stream>>>(Wq, Wqb, NW);
    k_cvt<<<1024, 256, 0, stream>>>(Wk, Wkb, NW);
    k_cvt<<<1024, 256, 0, stream>>>(Wv, Wvb, NW);
    k_cvt<<<1024, 256, 0, stream>>>(Wo, Wob, NW);

    // 2) projections (WMMA), head-split bf16 outputs
    dim3 ggrid(MTOT / 64, DMODEL / 128);
    k_gemm<0><<<ggrid, 256, 0, stream>>>(Qb, Wqb, qhb);
    k_gemm<0><<<ggrid, 256, 0, stream>>>(Kb, Wkb, khb);
    k_gemm<0><<<ggrid, 256, 0, stream>>>(Vb, Wvb, vhb);

    // 3) fused flash attention (WMMA + pipelined TDM + fp32 online softmax)
    k_flash<<<dim3(SLEN / 64, BATCH * HEADS), 128, 0, stream>>>(qhb, khb, vhb, ctx);

    // 4) output projection -> fp32 result
    k_gemm<1><<<ggrid, 256, 0, stream>>>(ctx, Wob, (float*)d_out);
}
